// LSTMLayer_6811818132101
// MI455X (gfx1250) — matrix operand order repaired and checked by reading, not compiled, not run on hardware
//
#include <hip/hip_runtime.h>
#include <hip/hip_bf16.h>

// Persistent mixed-precision LSTM for gfx1250 (MI455X).
// - v_wmma_f32_16x16x32_bf16 with fp32 accumulators
// - cell state resident in VGPRs across all 512 steps
// - h state ping-ponged through L2 in bf16
// - double-buffered LDS staging; h-chunks staged with
//   GLOBAL_LOAD_ASYNC_TO_LDS_B128 (ASYNCcnt) overlapped with WMMA
// - one grid-wide sync per timestep

#define KT   512      // timesteps
#define KB   64       // batch
#define KI   1024     // input dim
#define KH   1024     // hidden dim
#define K2   2048     // K = I + H for fused [x;h] GEMM
#define NWG  32       // persistent workgroups (each owns 32 hidden cols)
#define NTHR 256      // 8 wave32 per WG
#define ZSTR 264      // LDS row stride (bf16 elems), padded vs bank conflicts

typedef __attribute__((ext_vector_type(16))) __bf16 v16bf;
typedef __attribute__((ext_vector_type(8)))  __bf16 v8bf;
typedef __attribute__((ext_vector_type(8)))  float  v8f;
typedef __attribute__((ext_vector_type(4)))  float  v4f;

__device__ __forceinline__ __bf16 f2bf(float f) {
  unsigned u = __builtin_bit_cast(unsigned, f);
  u += 0x7FFFu + ((u >> 16) & 1u);            // round-to-nearest-even
  unsigned short h = (unsigned short)(u >> 16);
  return __builtin_bit_cast(__bf16, h);
}

__device__ __forceinline__ float sigmoidf_(float x) {
  return 1.0f / (1.0f + __expf(-x));
}

// ---- setup: pack W = [W_ih | W_hh] row-major [4H][I+H] in bf16 ----
__global__ void pack_weights_kernel(const float* __restrict__ Wih,
                                    const float* __restrict__ Whh,
                                    __bf16* __restrict__ Wp) {
  size_t idx = (size_t)blockIdx.x * NTHR + threadIdx.x;
  if (idx >= (size_t)4 * KH * K2) return;
  size_t r = idx >> 11;                // gate row 0..4095
  int    k = (int)(idx & (K2 - 1));    // 0..2047
  float v = (k < KI) ? Wih[r * KI + k] : Whh[r * KH + (k - KI)];
  Wp[idx] = f2bf(v);
}

// ---- setup: h0 -> bf16 ping buffer, fused bias, barrier reset ----
__global__ void init_state_kernel(const float* __restrict__ h0,
                                  const float* __restrict__ bih,
                                  const float* __restrict__ bhh,
                                  __bf16* __restrict__ hbuf,
                                  float* __restrict__ bias,
                                  unsigned* __restrict__ bar) {
  int idx = blockIdx.x * NTHR + threadIdx.x;
  if (idx < KB * KH) {
    hbuf[idx] = f2bf(h0[idx]);
  } else if (idx < KB * KH + 4 * KH) {
    int r = idx - KB * KH;
    bias[r] = bih[r] + bhh[r];
  } else if (idx < KB * KH + 4 * KH + 2) {
    bar[idx - KB * KH - 4 * KH] = 0u;
  }
}

// ---- device-wide barrier for the persistent kernel ----
__device__ __forceinline__ void grid_sync(unsigned* bar) {
  __threadfence();
  __syncthreads();
  if (threadIdx.x == 0) {
    unsigned* cnt = bar;
    unsigned* gen = bar + 1;
    unsigned g = __hip_atomic_load(gen, __ATOMIC_RELAXED, __HIP_MEMORY_SCOPE_AGENT);
    unsigned prev =
        __hip_atomic_fetch_add(cnt, 1u, __ATOMIC_ACQ_REL, __HIP_MEMORY_SCOPE_AGENT);
    if (prev + 1u == (unsigned)NWG) {
      __hip_atomic_store(cnt, 0u, __ATOMIC_RELAXED, __HIP_MEMORY_SCOPE_AGENT);
      __hip_atomic_fetch_add(gen, 1u, __ATOMIC_ACQ_REL, __HIP_MEMORY_SCOPE_AGENT);
    } else {
      while (__hip_atomic_load(gen, __ATOMIC_ACQUIRE, __HIP_MEMORY_SCOPE_AGENT) == g) {
        __builtin_amdgcn_s_sleep(2);
      }
    }
  }
  __threadfence();
  __syncthreads();
}

// ---- persistent recurrent kernel ----
// Wave w: M-tile mt = w&3 (16 batch rows), N-half nh = w>>2 (16 cols).
// Each wave holds 4 fp32 16x16 accumulators (gates i,f,g,o) + c tile in regs.
__global__ __launch_bounds__(NTHR) void lstm_persistent_kernel(
    const float* __restrict__ x, const float* __restrict__ c0,
    const __bf16* __restrict__ Wp, const float* __restrict__ bias,
    __bf16* __restrict__ hbuf, float* __restrict__ out,
    unsigned* __restrict__ bar) {
  // double-buffered K-chunk of z = [x_t ; h_t]
  __shared__ __align__(16) __bf16 zs[2][KB][ZSTR];

  const int tid  = threadIdx.x;
  const int lane = tid & 31;
  const int wave = tid >> 5;
  const int mt   = wave & 3;
  const int nh   = wave >> 2;
  const int sel  = lane >> 4;          // which half of the 16x16x32 K slice
  const int nn   = lane & 15;
  const int col  = blockIdx.x * 32 + nh * 16 + nn;   // hidden column 0..1023

  float bi[4];
#pragma unroll
  for (int g = 0; g < 4; ++g) bi[g] = bias[g * KH + col];

  v8f creg;                            // persistent cell state tile
#pragma unroll
  for (int r = 0; r < 8; ++r) {
    int row = mt * 16 + r + 8 * sel;   // C/D layout: VGPR r -> M = r + 8*sel
    creg[r] = c0[row * KH + col];
  }

  // stage one 256-wide K-chunk into LDS buffer bufi.
  // chunks 0..3: x_t (fp32 -> bf16 via VGPRs); chunks 4..7: h_t (bf16,
  // async global->LDS copy, tracked by ASYNCcnt, no VGPR round-trip).
  auto stage = [&](int t, int chunk, int bufi, const __bf16* hread) {
    for (int it = tid; it < (KB * 256) / 8; it += NTHR) {
      int row = it >> 5;
      int k8  = (it & 31) << 3;
      if (chunk < 4) {
        const float* s = x + ((size_t)t * KB + row) * KI + chunk * 256 + k8;
        v4f f0 = *(const v4f*)s;
        v4f f1 = *(const v4f*)(s + 4);
        v8bf v;
#pragma unroll
        for (int j = 0; j < 4; ++j) { v[j] = f2bf(f0[j]); v[4 + j] = f2bf(f1[j]); }
        *(v8bf*)&zs[bufi][row][k8] = v;
      } else {
        unsigned lds = (unsigned)(uintptr_t)&zs[bufi][row][k8];
        unsigned long long ga = (unsigned long long)(
            hread + (size_t)row * KH + (chunk - 4) * 256 + k8);
        asm volatile("global_load_async_to_lds_b128 %0, %1, off"
                     :: "v"(lds), "v"(ga) : "memory");
      }
    }
  };

  for (int t = 0; t < KT; ++t) {
    const __bf16* hread  = hbuf + (size_t)(t & 1) * (KB * KH);
    __bf16*       hwrite = hbuf + (size_t)((t + 1) & 1) * (KB * KH);

    v8f acc[4];
#pragma unroll
    for (int g = 0; g < 4; ++g)
#pragma unroll
      for (int r = 0; r < 8; ++r) acc[g][r] = bi[g];

    stage(t, 0, 0, hread);

    for (int chunk = 0; chunk < 8; ++chunk) {
      // drain this wave's outstanding async->LDS copies, then block barrier:
      // stage(chunk) is now visible to every wave.
      asm volatile("s_wait_asynccnt 0x0" ::: "memory");
      __syncthreads();

      // kick off next chunk's staging; its latency hides under the WMMAs below
      if (chunk < 7) stage(t, chunk + 1, (chunk + 1) & 1, hread);

      const int bufi  = chunk & 1;
      const int kbase = chunk * 256;

      // warm L2/WGP$ for next chunk's weight rows
      if (chunk < 7) {
#pragma unroll
        for (int g = 0; g < 4; ++g) {
          const __bf16* wpn = Wp + (size_t)(g * KH + col) * K2 + kbase + 256;
          __builtin_prefetch(wpn, 0, 1);
        }
      }

#pragma unroll
      for (int kk = 0; kk < 256; kk += 32) {
        // A fragment: rows of z for this wave's M-tile (from LDS)
        v8bf a0 = *(const v8bf*)&zs[bufi][mt * 16 + nn][kk + sel * 8];
        v8bf a1 = *(const v8bf*)&zs[bufi][mt * 16 + nn][kk + 16 + sel * 8];
        v16bf afrag = __builtin_shufflevector(
            a0, a1, 0, 1, 2, 3, 4, 5, 6, 7, 8, 9, 10, 11, 12, 13, 14, 15);
#pragma unroll
        for (int g = 0; g < 4; ++g) {
          // B fragment: weight row (gate g, column col), K = 8*sel+0..7 and 16+8*sel+0..7
          const __bf16* wp =
              Wp + (size_t)(g * KH + col) * K2 + kbase + kk + sel * 8;
          v8bf b0 = *(const v8bf*)wp;
          v8bf b1 = *(const v8bf*)(wp + 16);
          v16bf bfrag = __builtin_shufflevector(
              b0, b1, 0, 1, 2, 3, 4, 5, 6, 7, 8, 9, 10, 11, 12, 13, 14, 15);
          acc[g] = __builtin_amdgcn_wmma_f32_16x16x32_bf16(
              false, afrag, false, bfrag, (short)0, acc[g], false, false);
        }
      }
    }

    // ---- gates (torch order i,f,g,o), cell update, outputs ----
#pragma unroll
    for (int r = 0; r < 8; ++r) {
      int row  = mt * 16 + r + 8 * sel;
      float ig = sigmoidf_(acc[0][r]);
      float fg = sigmoidf_(acc[1][r]);
      float gg = tanhf(acc[2][r]);
      float og = sigmoidf_(acc[3][r]);
      float cn = fg * creg[r] + ig * gg;
      creg[r]  = cn;
      float hn = og * tanhf(cn);
      out[((size_t)t * KB + row) * KH + col] = hn;
      hwrite[(size_t)row * KH + col] = f2bf(hn);
      if (t == KT - 1)
        out[(size_t)KT * KB * KH + (size_t)row * KH + col] = hn;  // hT
    }
    grid_sync(bar);
  }

  // cT
#pragma unroll
  for (int r = 0; r < 8; ++r) {
    int row = mt * 16 + r + 8 * sel;
    out[(size_t)KT * KB * KH + (size_t)KB * KH + (size_t)row * KH + col] = creg[r];
  }
}

extern "C" void kernel_launch(void* const* d_in, const int* in_sizes, int n_in,
                              void* d_out, int out_size, void* d_ws, size_t ws_size,
                              hipStream_t stream) {
  (void)in_sizes; (void)n_in; (void)out_size; (void)ws_size;
  const float* x   = (const float*)d_in[0];
  const float* h0  = (const float*)d_in[1];
  const float* c0  = (const float*)d_in[2];
  const float* Wih = (const float*)d_in[3];
  const float* Whh = (const float*)d_in[4];
  const float* bih = (const float*)d_in[5];
  const float* bhh = (const float*)d_in[6];
  float* out = (float*)d_out;

  char* ws = (char*)d_ws;
  size_t off = 0;
  __bf16* Wp   = (__bf16*)(ws + off); off += (size_t)4 * KH * K2 * sizeof(__bf16); // 16 MB
  __bf16* hbuf = (__bf16*)(ws + off); off += (size_t)2 * KB * KH * sizeof(__bf16); // 256 KB
  float*  bias = (float*)(ws + off);  off += (size_t)4 * KH * sizeof(float);       // 16 KB
  unsigned* bar = (unsigned*)(ws + off);

  {
    size_t n = (size_t)4 * KH * K2;
    int blocks = (int)((n + NTHR - 1) / NTHR);
    pack_weights_kernel<<<blocks, NTHR, 0, stream>>>(Wih, Whh, Wp);
  }
  {
    int n = KB * KH + 4 * KH + 2;
    int blocks = (n + NTHR - 1) / NTHR;
    init_state_kernel<<<blocks, NTHR, 0, stream>>>(h0, bih, bhh, hbuf, bias, bar);
  }
  lstm_persistent_kernel<<<NWG, NTHR, 0, stream>>>(x, c0, Wp, bias, hbuf, out, bar);
}
